// MPNN_72507637891551
// MI455X (gfx1250) — matrix operand order repaired and checked by reading, not compiled, not run on hardware
//
#include <hip/hip_runtime.h>
#include <hip/hip_bf16.h>
#include <math.h>

typedef __bf16 bf16_t;
typedef __attribute__((ext_vector_type(16))) __bf16 v16bf;
typedef __attribute__((ext_vector_type(8)))  float  v8f;

#define N_NODES 20000
#define N_EDGES 200000
#define D_IN 32
#define D_H 32
#define D_OUT 8
#define E_DIM 16

__device__ __forceinline__ float elu_f(float v) { return v > 0.f ? v : (__expf(v) - 1.f); }

// Packed bf16 multiply (2 lanes per dword) — VOP3P op the compiler won't emit from C.
__device__ __forceinline__ unsigned int pk_mul_bf16(unsigned int a, unsigned int b) {
    unsigned int d;
    asm("v_pk_mul_bf16 %0, %1, %2" : "=v"(d) : "v"(a), "v"(b));
    return d;
}

__global__ void zero_i32(int* __restrict__ p, int n) {
    int i = blockIdx.x * blockDim.x + threadIdx.x;
    int stride = gridDim.x * blockDim.x;
    for (; i < n; i += stride) p[i] = 0;
}

__global__ void degree_kernel(const int* __restrict__ dst, int* __restrict__ deg, int ne) {
    int e = blockIdx.x * blockDim.x + threadIdx.x;
    if (e < ne) atomicAdd(&deg[dst[e]], 1);
}

// Single-block exclusive scan of deg -> off[0..n], cursor copy, inv_count.
__global__ void scan_kernel(const int* __restrict__ deg, int* __restrict__ off,
                            int* __restrict__ cursor, float* __restrict__ inv, int n) {
    __shared__ int partial[256];
    int tid = threadIdx.x;
    int nthr = blockDim.x;
    int chunk = (n + nthr - 1) / nthr;
    int begin = tid * chunk;
    int end = begin + chunk < n ? begin + chunk : n;
    int s = 0;
    for (int v = begin; v < end; ++v) s += deg[v];
    partial[tid] = s;
    __syncthreads();
    // Hillis-Steele inclusive scan over partials
    for (int d = 1; d < 256; d <<= 1) {
        int t = (tid >= d) ? partial[tid - d] : 0;
        __syncthreads();
        partial[tid] += t;
        __syncthreads();
    }
    int run = (tid == 0) ? 0 : partial[tid - 1];
    for (int v = begin; v < end; ++v) {
        off[v] = run;
        cursor[v] = run;
        int d = deg[v];
        inv[v] = 1.0f / fmaxf((float)d, 1.0f);
        run += d;
    }
    if (tid == nthr - 1) off[n] = run;
}

// slot[e] = unique position within dst[e]'s contiguous segment
__global__ void slot_kernel(const int* __restrict__ dst, int* __restrict__ cursor,
                            int* __restrict__ slot, int ne) {
    int e = blockIdx.x * blockDim.x + threadIdx.x;
    if (e < ne) slot[e] = atomicAdd(&cursor[dst[e]], 1);
}

// h = elu(edge_attr @ w1 + b1), stored bf16 [E,16]
__global__ void edge_mlp_kernel(const float* __restrict__ ea,
                                const float* __restrict__ w1,
                                const float* __restrict__ b1,
                                bf16_t* __restrict__ hout, int ne) {
    __shared__ float w1s[E_DIM * E_DIM];
    __shared__ float b1s[E_DIM];
    for (int i = threadIdx.x; i < E_DIM * E_DIM; i += blockDim.x) w1s[i] = w1[i];
    if (threadIdx.x < E_DIM) b1s[threadIdx.x] = b1[threadIdx.x];
    __syncthreads();
    int e = blockIdx.x * blockDim.x + threadIdx.x;
    if (e >= ne) return;
    const float* arow = ea + (size_t)e * E_DIM;
    float a[E_DIM];
#pragma unroll
    for (int k = 0; k < E_DIM; ++k) a[k] = arow[k];
    bf16_t outv[E_DIM];
#pragma unroll
    for (int j = 0; j < E_DIM; ++j) {
        float s = b1s[j];
#pragma unroll
        for (int k = 0; k < E_DIM; ++k) s = fmaf(a[k], w1s[k * E_DIM + j], s);
        outv[j] = (bf16_t)elu_f(s);
    }
    uint4* dp = (uint4*)(hout + (size_t)e * E_DIM);
    dp[0] = ((const uint4*)outv)[0];
    dp[1] = ((const uint4*)outv)[1];
}

// Fused msg GEMM + sorted-slot store:  msg = Z(E x 544) @ W2'(544 x 32).
// Z[e, t*32+i] = h[e,t]*x_src[e,i] for t<16;  Z[e, 512+i] = x_src[e,i] (b2 rows).
// One wave per 16-edge tile; 17 K-steps x 2 N-tiles of v_wmma_f32_16x16x32_bf16.
// Messages land at msgbuf[slot[e]] so aggregation is a contiguous, atomic-free sum.
__global__ void __launch_bounds__(256)
nnconv_msg_kernel(const float* __restrict__ xin,   // [N,32]
                  const bf16_t* __restrict__ hbf,  // [E,16]
                  const float* __restrict__ w2,    // [16,1024] = w2[k, i*32+o]
                  const float* __restrict__ b2,    // [1024]    = b2[i*32+o]
                  const int* __restrict__ src,
                  const int* __restrict__ slot,
                  float* __restrict__ msgbuf,      // [E,32]
                  int ne) {
    // B operand staged as w2t[k][o][i] bf16 so each B reg-set is 32 contiguous bytes.
    __shared__ bf16_t w2t[17][32][32];  // 34 KB
    for (int idx = threadIdx.x; idx < 16 * 1024; idx += blockDim.x) {
        int k = idx >> 10, rem = idx & 1023, i = rem >> 5, o = rem & 31;
        w2t[k][o][i] = (bf16_t)w2[idx];
    }
    for (int idx = threadIdx.x; idx < 1024; idx += blockDim.x) {
        int i = idx >> 5, o = idx & 31;
        w2t[16][o][i] = (bf16_t)b2[idx];
    }
    __syncthreads();

    const int lane = threadIdx.x & 31;
    const int lm   = lane & 15;       // row (edge in tile) for A; column for B/C
    const int half = lane >> 4;       // lane half selects K sub-range (contiguous k = 16*half + s, same for A and B)
    const int i0   = half * 16;       // A: K in {i0..i0+15}
    const int hb   = half * 16;       // B: K in {hb..hb+15}

    int gwave  = blockIdx.x * (blockDim.x >> 5) + (threadIdx.x >> 5);
    int nwave  = gridDim.x * (blockDim.x >> 5);
    int ntiles = (ne + 15) >> 4;

    for (int tile = gwave; tile < ntiles; tile += nwave) {
        int em = tile * 16 + lm;
        bool valid = em < ne;
        int s = valid ? src[em] : 0;
        const float* xr = xin + (size_t)s * 32;

        // x_src slice in A "idx" order: i=i0+idx (contiguous, matches B)
        float xf[16];
#pragma unroll
        for (int q = 0; q < 8; ++q) xf[q] = xr[i0 + q];
#pragma unroll
        for (int q = 0; q < 8; ++q) xf[8 + q] = xr[i0 + 8 + q];
        if (!valid) {
#pragma unroll
            for (int q = 0; q < 16; ++q) xf[q] = 0.f;
        }

        // h row as 8 dwords of packed bf16 pairs
        unsigned int hw[8];
        const unsigned int* hp = (const unsigned int*)(hbf + (size_t)em * 16);
#pragma unroll
        for (int q = 0; q < 8; ++q) hw[q] = valid ? hp[q] : 0u;

        union { v16bf v; unsigned int u[8]; } xu, au;
#pragma unroll
        for (int q = 0; q < 16; ++q) xu.v[q] = (bf16_t)xf[q];

        v8f acc0 = {};  // columns 0..15
        v8f acc1 = {};  // columns 16..31

#pragma unroll
        for (int t = 0; t < 17; ++t) {
            v16bf az;
            if (t < 16) {
                // splat h[m,t] (bf16) to both halves of a dword, then packed bf16 muls
                unsigned int w = hw[t >> 1];
                unsigned int hs = (t & 1) ? ((w & 0xffff0000u) | (w >> 16))
                                          : ((w << 16) | (w & 0x0000ffffu));
#pragma unroll
                for (int q = 0; q < 8; ++q) au.u[q] = pk_mul_bf16(xu.u[q], hs);
                az = au.v;
            } else {
                az = xu.v;  // bias-row block: coefficient 1, B rows = b2
            }
            v16bf bv0 = *(const v16bf*)&w2t[t][lm][hb];
            v16bf bv1 = *(const v16bf*)&w2t[t][lm + 16][hb];
            acc0 = __builtin_amdgcn_wmma_f32_16x16x32_bf16(false, az, false, bv0,
                                                           (short)0, acc0, false, false);
            acc1 = __builtin_amdgcn_wmma_f32_16x16x32_bf16(false, az, false, bv1,
                                                           (short)0, acc1, false, false);
        }

        // Store rows to sorted slots. C/D layout: VGPR j = row m=j+8*half, col lane%16.
        int sl = valid ? slot[em] : -1;
#pragma unroll
        for (int j = 0; j < 8; ++j) {
            int m  = j + 8 * half;
            int sv = __shfl(sl, m, 32);  // lane m (<16) holds slot[tile*16+m]
            if (sv >= 0) {
                float* mr = msgbuf + (size_t)sv * 32;
                mr[lm]      = acc0[j];
                mr[16 + lm] = acc1[j];
            }
        }
    }
}

// out = elu( sum(msg slots)/deg + x@root + bias ) — contiguous, atomic-free aggregation
__global__ void node_update_kernel(const int* __restrict__ off,
                                   const float* __restrict__ msgbuf,
                                   const float* __restrict__ invc,
                                   const float* __restrict__ xin,
                                   const float* __restrict__ root,  // [32,32]
                                   const float* __restrict__ bias,  // [32]
                                   float* __restrict__ outp, int n) {
    __shared__ float rs[32 * 32];
    __shared__ float bs[32];
    for (int i = threadIdx.x; i < 1024; i += blockDim.x) rs[i] = root[i];
    if (threadIdx.x < 32) bs[threadIdx.x] = bias[threadIdx.x];
    __syncthreads();
    int v = blockIdx.x * blockDim.x + threadIdx.x;
    if (v >= n) return;

    float accv[32];
#pragma unroll
    for (int i = 0; i < 32; ++i) accv[i] = 0.f;
    int b = off[v], e = off[v + 1];
    for (int p = b; p < e; ++p) {
        const float* mr = msgbuf + (size_t)p * 32;
#pragma unroll
        for (int i = 0; i < 32; ++i) accv[i] += mr[i];
    }

    const float* xr = xin + (size_t)v * 32;
    float xv[32];
#pragma unroll
    for (int i = 0; i < 32; ++i) xv[i] = xr[i];
    float ic = invc[v];
    float* orow = outp + (size_t)v * 32;
#pragma unroll
    for (int o = 0; o < 32; ++o) {
        float s = fmaf(accv[o], ic, bs[o]);
#pragma unroll
        for (int i = 0; i < 32; ++i) s = fmaf(xv[i], rs[i * 32 + o], s);
        orow[o] = elu_f(s);
    }
}

__global__ void classifier_kernel(const float* __restrict__ h,
                                  const float* __restrict__ w,  // [32,8]
                                  const float* __restrict__ b,  // [8]
                                  float* __restrict__ outp, int n) {
    __shared__ float wsm[32 * D_OUT];
    __shared__ float bsm[D_OUT];
    for (int i = threadIdx.x; i < 32 * D_OUT; i += blockDim.x) wsm[i] = w[i];
    if (threadIdx.x < D_OUT) bsm[threadIdx.x] = b[threadIdx.x];
    __syncthreads();
    int v = blockIdx.x * blockDim.x + threadIdx.x;
    if (v >= n) return;
    const float* hr = h + (size_t)v * 32;
    float xv[32];
#pragma unroll
    for (int i = 0; i < 32; ++i) xv[i] = hr[i];
#pragma unroll
    for (int o = 0; o < D_OUT; ++o) {
        float s = bsm[o];
#pragma unroll
        for (int i = 0; i < 32; ++i) s = fmaf(xv[i], wsm[i * D_OUT + o], s);
        outp[(size_t)v * D_OUT + o] = s;
    }
}

extern "C" void kernel_launch(void* const* d_in, const int* in_sizes, int n_in,
                              void* d_out, int out_size, void* d_ws, size_t ws_size,
                              hipStream_t stream) {
    (void)in_sizes; (void)n_in; (void)out_size; (void)ws_size;
    const float* x         = (const float*)d_in[0];
    const float* edge_attr = (const float*)d_in[1];
    const int*   edge_idx  = (const int*)d_in[2];
    const float* w1_0 = (const float*)d_in[3];
    const float* b1_0 = (const float*)d_in[4];
    const float* w2_0 = (const float*)d_in[5];
    const float* b2_0 = (const float*)d_in[6];
    const float* root_0 = (const float*)d_in[7];
    const float* bias_0 = (const float*)d_in[8];
    const float* w1_1 = (const float*)d_in[9];
    const float* b1_1 = (const float*)d_in[10];
    const float* w2_1 = (const float*)d_in[11];
    const float* b2_1 = (const float*)d_in[12];
    const float* root_1 = (const float*)d_in[13];
    const float* bias_1 = (const float*)d_in[14];
    const float* cls_w  = (const float*)d_in[15];
    const float* cls_b  = (const float*)d_in[16];
    float* out = (float*)d_out;

    const int* src = edge_idx;            // edge_index[0]
    const int* dst = edge_idx + N_EDGES;  // edge_index[1]

    char* wsp = (char*)d_ws;
    auto take = [&](size_t bytes) {
        char* p = wsp;
        wsp += (bytes + 255) & ~(size_t)255;
        return p;
    };
    int*    deg    = (int*)take((size_t)N_NODES * 4);
    int*    off    = (int*)take((size_t)(N_NODES + 1) * 4);
    int*    cursor = (int*)take((size_t)N_NODES * 4);
    int*    slot   = (int*)take((size_t)N_EDGES * 4);
    float*  inv    = (float*)take((size_t)N_NODES * 4);
    float*  msgbuf = (float*)take((size_t)N_EDGES * 32 * 4);  // 25.6 MB (L2-resident)
    float*  h1     = (float*)take((size_t)N_NODES * 32 * 4);
    float*  h2     = (float*)take((size_t)N_NODES * 32 * 4);
    bf16_t* hbf    = (bf16_t*)take((size_t)N_EDGES * 16 * 2);

    const int TB = 256;
    // CSR-style sort-by-dst metadata (done once; dst is identical for both layers)
    zero_i32<<<128, TB, 0, stream>>>(deg, N_NODES);
    degree_kernel<<<(N_EDGES + TB - 1) / TB, TB, 0, stream>>>(dst, deg, N_EDGES);
    scan_kernel<<<1, TB, 0, stream>>>(deg, off, cursor, inv, N_NODES);
    slot_kernel<<<(N_EDGES + TB - 1) / TB, TB, 0, stream>>>(dst, cursor, slot, N_EDGES);

    // layer 0
    edge_mlp_kernel<<<(N_EDGES + TB - 1) / TB, TB, 0, stream>>>(edge_attr, w1_0, b1_0, hbf, N_EDGES);
    nnconv_msg_kernel<<<256, TB, 0, stream>>>(x, hbf, w2_0, b2_0, src, slot, msgbuf, N_EDGES);
    node_update_kernel<<<(N_NODES + TB - 1) / TB, TB, 0, stream>>>(off, msgbuf, inv, x, root_0, bias_0, h1, N_NODES);

    // layer 1
    edge_mlp_kernel<<<(N_EDGES + TB - 1) / TB, TB, 0, stream>>>(edge_attr, w1_1, b1_1, hbf, N_EDGES);
    nnconv_msg_kernel<<<256, TB, 0, stream>>>(h1, hbf, w2_1, b2_1, src, slot, msgbuf, N_EDGES);
    node_update_kernel<<<(N_NODES + TB - 1) / TB, TB, 0, stream>>>(off, msgbuf, inv, h1, root_1, bias_1, h2, N_NODES);

    // classifier
    classifier_kernel<<<(N_NODES + TB - 1) / TB, TB, 0, stream>>>(h2, cls_w, cls_b, out, N_NODES);
}
